// BBoxVAE_45457933861470
// MI455X (gfx1250) — matrix operand order repaired and checked by reading, not compiled, not run on hardware
//
#include <hip/hip_runtime.h>
#include <hip/hip_bf16.h>

typedef _Float16 h16;
typedef __attribute__((ext_vector_type(16))) _Float16 v16h;
typedef __attribute__((ext_vector_type(8)))  float    v8f;
typedef __attribute__((ext_vector_type(4)))  unsigned int v4u;

#define N_LABELS 24
#define NSTEP    24

// ---------------- workspace layout (f16 halves) ----------------
#define OFF_CPB   0        // c_pb_W            [128][128]
#define OFF_C3    16384    // c_out_W rows 256..383 [128][128]
#define OFF_E1P   32768    // e1_W K-pad        [32][128]
#define OFF_E2    36864    // e2_W              [128][128]
#define OFF_E3    53248    // e3_W              [256][32]
#define OFF_ML    61440    // e_mu_W|e_lv_W     [32][64]
#define OFF_D1    63488    // d1_W              [160][128]
#define OFF_D2    83968    // d2_W              [128][64]
#define OFF_D3P   92160    // d3_W N-pad        [64][16]
#define OFF_L4P   93184    // lstm_W rows 24..27 K-pad [32][512]
#define OFF_U     109568   // lstm_U            [128][512]
#define OFF_LS1P  175104   // c_ls1_W K-pad     [32][128]
#define OFF_LS2   179200   // c_ls2_W           [128][128]
#define OFF_CA    195584   // c_out_W rows 0..127 [128][128]
#define TOT_H16   211968
// f32 region
#define OFF_STEPC_BYTES (TOT_H16*2)                    // [24][128]
#define OFF_STEPL_BYTES (OFF_STEPC_BYTES + 24*128*4)   // [24][512]
#define OFF_ZB_BYTES    (OFF_STEPL_BYTES + 24*512*4)   // [128] zeros

// ---------------- LDS staging: one wave handles 32 rows (2 M-tiles) --------
#define SP   144   // stride (halves) for 128-col buffers
#define SIZ  80    // stride for 64-col buffers
#define SBB  32    // stride for bbox buffer
#define L_PREV  0        // 32*144
#define L_CBASE 4608     // 32*144
#define L_COND  9216     // 32*144
#define L_BUF   13824    // 32*144
#define L_IZ    18432    // 32*80
#define L_H2    20992    // 32*80
#define L_BB    23552    // 32*32
#define L_TOT   24576    // 48 KB

union H16x16 { v16h v; v4u u[2]; h16 h[16]; };

__device__ __forceinline__ v16h a_frag(const h16* S, int stride, int kb, int lane) {
    int rr = lane & 15, hh = (lane >> 4) & 1;
    const h16* p = S + rr * stride + kb + hh * 8;
    H16x16 t;
    t.u[0] = *(const v4u*)(p);
    t.u[1] = *(const v4u*)(p + 16);
    return t.v;
}

__device__ __forceinline__ v16h b_frag(const h16* W, int ldn, int kb, int nb, int lane) {
    const h16* p = W + (size_t)kb * ldn + nb + (lane & 15);
    int hh = (lane >> 4) & 1;
    H16x16 t;
#pragma unroll
    for (int s = 0; s < 16; ++s) {
        int k = (s < 8) ? (8 * hh + s) : (16 + 8 * hh + (s - 8));
        t.h[s] = p[(size_t)k * ldn];
    }
    return t.v;
}

__device__ __forceinline__ v8f wmma32(v16h a, v16h b, v8f c) {
    return __builtin_amdgcn_wmma_f32_16x16x32_f16(false, a, false, b, (short)0, c,
                                                  false, false);
}

// bias must always be a valid pointer (use zero vector for "no bias")
__device__ __forceinline__ void c_store16(h16* S, int stride, int nb, int lane,
                                          v8f acc, const float* bias, int do_relu) {
    int col = nb + (lane & 15);
    int rb  = (lane >> 4) * 8;
    float b = bias[col];
#pragma unroll
    for (int v = 0; v < 8; ++v) {
        float x = acc[v] + b;
        if (do_relu) x = fmaxf(x, 0.f);
        S[(rb + v) * stride + col] = (h16)x;
    }
}

__device__ __forceinline__ float sigm(float x) { return 1.f / (1.f + __expf(-x)); }

// ---------------- prep kernels ----------------
__global__ void zero16_kernel(h16* dst, int n) {
    int i = blockIdx.x * blockDim.x + threadIdx.x;
    if (i < n) dst[i] = (h16)0.f;
}

__global__ void conv_kernel(h16* __restrict__ dst, int ld_dst, int dr0, int dc0,
                            const float* __restrict__ src, int ld_src, int sr0, int sc0,
                            int rows, int cols) {
    int i = blockIdx.x * blockDim.x + threadIdx.x;
    if (i >= rows * cols) return;
    int rr = i / cols, cc = i % cols;
    dst[(size_t)(dr0 + rr) * ld_dst + dc0 + cc] =
        (h16)src[(size_t)(sr0 + rr) * ld_src + sc0 + cc];
}

__global__ __launch_bounds__(128) void step_consts_kernel(
    const float* __restrict__ c_cl1_W, const float* __restrict__ c_cl1_b,
    const float* __restrict__ c_cl2_W, const float* __restrict__ c_cl2_b,
    const float* __restrict__ c_out_W, const float* __restrict__ c_out_b,
    const float* __restrict__ lstm_W,  const float* __restrict__ lstm_b,
    float* __restrict__ stepc, float* __restrict__ stepl, float* __restrict__ zb) {
    int t = blockIdx.x, n = threadIdx.x;
    __shared__ float a1[128], a2[128];
    if (t == 0) zb[n] = 0.f;
    a1[n] = fmaxf(c_cl1_W[t * 128 + n] + c_cl1_b[n], 0.f);
    __syncthreads();
    float s = 0.f;
    for (int k = 0; k < 128; ++k) s += a1[k] * c_cl2_W[k * 128 + n];
    a2[n] = fmaxf(s + c_cl2_b[n], 0.f);
    __syncthreads();
    s = 0.f;
    for (int k = 0; k < 128; ++k) s += a2[k] * c_out_W[(128 + k) * 128 + n];
    stepc[t * 128 + n] = s + c_out_b[n];
    for (int g = 0; g < 4; ++g) {
        int j = g * 128 + n;
        stepl[t * 512 + j] = lstm_W[t * 512 + j] + lstm_b[j];
    }
}

// ---------------- main kernel: one wave handles 32 batch rows (2 M-tiles) ----
__global__ __launch_bounds__(32)
__attribute__((amdgpu_waves_per_eu(2)))
void bboxvae_main(
    const float* __restrict__ label_set, const float* __restrict__ bbox_in,
    const float* __restrict__ eps, const h16* __restrict__ ws16,
    const float* __restrict__ stepc, const float* __restrict__ stepl,
    const float* __restrict__ zb,
    const float* __restrict__ b_ls1, const float* __restrict__ b_ls2,
    const float* __restrict__ b_cpb, const float* __restrict__ b_e1,
    const float* __restrict__ b_e2,  const float* __restrict__ b_e3,
    const float* __restrict__ b_mu,  const float* __restrict__ b_lv,
    const float* __restrict__ b_d1,  const float* __restrict__ b_d2,
    const float* __restrict__ b_d3,  float* __restrict__ out) {
    __shared__ __align__(16) h16 smem[L_TOT];
    h16* sc_prev  = smem + L_PREV;
    h16* sc_cbase = smem + L_CBASE;
    h16* sc_cond  = smem + L_COND;
    h16* sc_buf   = smem + L_BUF;
    h16* sc_iz    = smem + L_IZ;
    h16* sc_h2    = smem + L_H2;
    h16* sc_bbox  = smem + L_BB;

    const int lane = threadIdx.x;
    const int row0 = blockIdx.x * 32;
    const int r    = lane & 15;
    const int hi   = lane >> 4;
    const int r0   = hi * 8;      // C-layout row base within tile
    const int c15  = lane & 15;   // C-layout column within tile
    const v8f Z8 = {0.f, 0.f, 0.f, 0.f, 0.f, 0.f, 0.f, 0.f};

    {   // zero all staging (prev state = 0, bbox pad cols = 0, no NaN garbage)
        unsigned int* p = (unsigned int*)smem;
        for (int i = lane; i < L_TOT / 2; i += 32) p[i] = 0u;
    }
    __syncthreads();

    // ---- prologue: cond_base = relu(relu(ls@Wls1+b)@Wls2+b) @ WcA ----
    {
        v16h lsf[2];
#pragma unroll
        for (int m = 0; m < 2; ++m) {
            float lsv[24];
            const float* lsrow = label_set + (size_t)(row0 + m * 16 + r) * N_LABELS;
#pragma unroll
            for (int q = 0; q < 6; ++q) {
                float4 f4 = *(const float4*)(lsrow + q * 4);
                lsv[q * 4 + 0] = f4.x; lsv[q * 4 + 1] = f4.y;
                lsv[q * 4 + 2] = f4.z; lsv[q * 4 + 3] = f4.w;
            }
            H16x16 u;
#pragma unroll
            for (int h = 0; h < 16; ++h) {
                int k0 = h + 8 * (h >> 3);                       // 0..7, 16..23
                float v0 = lsv[k0];
                float v1 = (k0 + 8 < N_LABELS) ? lsv[k0 + 8] : 0.f;
                u.h[h] = (h16)(hi ? v1 : v0);                    // branchless select
            }
            lsf[m] = u.v;
        }
#pragma unroll 2
        for (int n = 0; n < 8; ++n) {
            v8f acc[2] = {Z8, Z8};
            v16h b = b_frag(ws16 + OFF_LS1P, 128, 0, n * 16, lane);
#pragma unroll
            for (int m = 0; m < 2; ++m) acc[m] = wmma32(lsf[m], b, acc[m]);
#pragma unroll
            for (int m = 0; m < 2; ++m)
                c_store16(sc_buf + m * 16 * SP, SP, n * 16, lane, acc[m], b_ls1, 1);
        }
    }
    __syncthreads();
    {
        v16h af[2][4];
#pragma unroll
        for (int m = 0; m < 2; ++m)
#pragma unroll
            for (int k = 0; k < 4; ++k)
                af[m][k] = a_frag(sc_buf + m * 16 * SP, SP, 32 * k, lane);
#pragma unroll 2
        for (int n = 0; n < 8; ++n) {
            v8f acc[2] = {Z8, Z8};
#pragma unroll
            for (int k = 0; k < 4; ++k) {
                v16h b = b_frag(ws16 + OFF_LS2, 128, 32 * k, n * 16, lane);
                acc[0] = wmma32(af[0][k], b, acc[0]);
                acc[1] = wmma32(af[1][k], b, acc[1]);
            }
#pragma unroll
            for (int m = 0; m < 2; ++m)
                c_store16(sc_buf + m * 16 * SP, SP, n * 16, lane, acc[m], b_ls2, 1);
        }
    }
    __syncthreads();
    {
        v16h af[2][4];
#pragma unroll
        for (int m = 0; m < 2; ++m)
#pragma unroll
            for (int k = 0; k < 4; ++k)
                af[m][k] = a_frag(sc_buf + m * 16 * SP, SP, 32 * k, lane);
#pragma unroll 2
        for (int n = 0; n < 8; ++n) {
            v8f acc[2] = {Z8, Z8};
#pragma unroll
            for (int k = 0; k < 4; ++k) {
                v16h b = b_frag(ws16 + OFF_CA, 128, 32 * k, n * 16, lane);
                acc[0] = wmma32(af[0][k], b, acc[0]);
                acc[1] = wmma32(af[1][k], b, acc[1]);
            }
#pragma unroll
            for (int m = 0; m < 2; ++m)
                c_store16(sc_cbase + m * 16 * SP, SP, n * 16, lane, acc[m], zb, 0);
        }
    }
    __syncthreads();

    // ---- recurrent step loop ----
#pragma unroll 1
    for (int t = 0; t < NSTEP; ++t) {
        // i3 = relu(prev @ c_pb_W + b)
        {
            v16h af[2][4];
#pragma unroll
            for (int m = 0; m < 2; ++m)
#pragma unroll
                for (int k = 0; k < 4; ++k)
                    af[m][k] = a_frag(sc_prev + m * 16 * SP, SP, 32 * k, lane);
#pragma unroll 2
            for (int n = 0; n < 8; ++n) {
                v8f acc[2] = {Z8, Z8};
#pragma unroll
                for (int k = 0; k < 4; ++k) {
                    v16h b = b_frag(ws16 + OFF_CPB, 128, 32 * k, n * 16, lane);
                    acc[0] = wmma32(af[0][k], b, acc[0]);
                    acc[1] = wmma32(af[1][k], b, acc[1]);
                }
#pragma unroll
                for (int m = 0; m < 2; ++m)
                    c_store16(sc_buf + m * 16 * SP, SP, n * 16, lane, acc[m], b_cpb, 1);
            }
        }
        __syncthreads();

        // cond = cond_base + stepcond[t] + i3 @ c_out_W[256:384]
        {
            v16h af[2][4];
#pragma unroll
            for (int m = 0; m < 2; ++m)
#pragma unroll
                for (int k = 0; k < 4; ++k)
                    af[m][k] = a_frag(sc_buf + m * 16 * SP, SP, 32 * k, lane);
#pragma unroll 2
            for (int n = 0; n < 8; ++n) {
                v8f acc[2] = {Z8, Z8};
#pragma unroll
                for (int k = 0; k < 4; ++k) {
                    v16h b = b_frag(ws16 + OFF_C3, 128, 32 * k, n * 16, lane);
                    acc[0] = wmma32(af[0][k], b, acc[0]);
                    acc[1] = wmma32(af[1][k], b, acc[1]);
                }
                int col = n * 16 + c15;
                float sb = stepc[t * 128 + col];
#pragma unroll
                for (int m = 0; m < 2; ++m) {
                    const h16* cb = sc_cbase + m * 16 * SP;
                    h16* cd = sc_cond + m * 16 * SP;
#pragma unroll
                    for (int v = 0; v < 8; ++v) {
                        float x = acc[m][v] + sb + (float)cb[(r0 + v) * SP + col];
                        cd[(r0 + v) * SP + col] = (h16)x;
                    }
                }
            }
        }
        __syncthreads();

        // encoder: h = relu(gt@e1+b) @ e2 + b
        {
            v16h gf[2];
#pragma unroll
            for (int m = 0; m < 2; ++m) {
                float4 g = *(const float4*)(bbox_in +
                    ((size_t)(row0 + m * 16 + r) * 24 + t) * 4);
                float gv[4] = {g.x, g.y, g.z, g.w};
                H16x16 u;
#pragma unroll
                for (int h = 0; h < 16; ++h) {
                    int k0 = h + 8 * (h >> 3);
                    float v0 = (k0 < 4) ? gv[k0 < 4 ? k0 : 0] : 0.f;  // compile-time
                    u.h[h] = (h16)(hi ? 0.f : v0);                    // branchless
                }
                gf[m] = u.v;
            }
#pragma unroll 2
            for (int n = 0; n < 8; ++n) {
                v8f acc[2] = {Z8, Z8};
                v16h b = b_frag(ws16 + OFF_E1P, 128, 0, n * 16, lane);
#pragma unroll
                for (int m = 0; m < 2; ++m) acc[m] = wmma32(gf[m], b, acc[m]);
#pragma unroll
                for (int m = 0; m < 2; ++m)
                    c_store16(sc_buf + m * 16 * SP, SP, n * 16, lane, acc[m], b_e1, 1);
            }
        }
        __syncthreads();
        {
            v16h af[2][4];
#pragma unroll
            for (int m = 0; m < 2; ++m)
#pragma unroll
                for (int k = 0; k < 4; ++k)
                    af[m][k] = a_frag(sc_buf + m * 16 * SP, SP, 32 * k, lane);
#pragma unroll 2
            for (int n = 0; n < 8; ++n) {
                v8f acc[2] = {Z8, Z8};
#pragma unroll
                for (int k = 0; k < 4; ++k) {
                    v16h b = b_frag(ws16 + OFF_E2, 128, 32 * k, n * 16, lane);
                    acc[0] = wmma32(af[0][k], b, acc[0]);
                    acc[1] = wmma32(af[1][k], b, acc[1]);
                }
#pragma unroll
                for (int m = 0; m < 2; ++m)   // no relu on h
                    c_store16(sc_buf + m * 16 * SP, SP, n * 16, lane, acc[m], b_e2, 0);
            }
        }
        __syncthreads();

        // inter = relu(concat(h, cond) @ e3 + b)
        {
            v16h ah[2][4], ac[2][4];
#pragma unroll
            for (int m = 0; m < 2; ++m)
#pragma unroll
                for (int k = 0; k < 4; ++k) {
                    ah[m][k] = a_frag(sc_buf + m * 16 * SP, SP, 32 * k, lane);
                    ac[m][k] = a_frag(sc_cond + m * 16 * SP, SP, 32 * k, lane);
                }
#pragma unroll
            for (int n = 0; n < 2; ++n) {
                v8f acc[2] = {Z8, Z8};
#pragma unroll
                for (int k = 0; k < 4; ++k) {
                    v16h b = b_frag(ws16 + OFF_E3, 32, 32 * k, n * 16, lane);
                    acc[0] = wmma32(ah[0][k], b, acc[0]);
                    acc[1] = wmma32(ah[1][k], b, acc[1]);
                }
#pragma unroll
                for (int k = 0; k < 4; ++k) {
                    v16h b = b_frag(ws16 + OFF_E3, 32, 128 + 32 * k, n * 16, lane);
                    acc[0] = wmma32(ac[0][k], b, acc[0]);
                    acc[1] = wmma32(ac[1][k], b, acc[1]);
                }
#pragma unroll
                for (int m = 0; m < 2; ++m)
                    c_store16(sc_iz + m * 16 * SIZ, SIZ, n * 16, lane, acc[m], b_e3, 1);
            }
        }
        __syncthreads();

        // mu / logvar fused + reparam sample
        {
            v16h itf[2];
#pragma unroll
            for (int m = 0; m < 2; ++m) itf[m] = a_frag(sc_iz + m * 16 * SIZ, SIZ, 0, lane);
            v8f am[2][2], al[2][2];
#pragma unroll
            for (int n = 0; n < 2; ++n) {
                v16h bm = b_frag(ws16 + OFF_ML, 64, 0, n * 16, lane);
                v16h bl = b_frag(ws16 + OFF_ML, 64, 0, 32 + n * 16, lane);
#pragma unroll
                for (int m = 0; m < 2; ++m) {
                    am[m][n] = wmma32(itf[m], bm, Z8);
                    al[m][n] = wmma32(itf[m], bl, Z8);
                }
            }
#pragma unroll
            for (int n = 0; n < 2; ++n) {
                int col = n * 16 + c15;
                float bmv = b_mu[col], blv = b_lv[col];
#pragma unroll
                for (int m = 0; m < 2; ++m) {
                    h16* zs = sc_iz + m * 16 * SIZ;
#pragma unroll
                    for (int v = 0; v < 8; ++v) {
                        int grow = row0 + m * 16 + r0 + v;
                        float mu = am[m][n][v] + bmv;
                        float lv = al[m][n][v] + blv;
                        float e  = eps[((size_t)grow * 24 + t) * 32 + col];
                        float z  = mu + __expf(0.5f * lv) * e;
                        zs[(r0 + v) * SIZ + 32 + col] = (h16)z;
                    }
                }
            }
        }
        __syncthreads();

        // decoder d1: relu(concat(cond, z) @ d1 + b)
        {
            v16h ac[2][4], zfr[2];
#pragma unroll
            for (int m = 0; m < 2; ++m) {
#pragma unroll
                for (int k = 0; k < 4; ++k)
                    ac[m][k] = a_frag(sc_cond + m * 16 * SP, SP, 32 * k, lane);
                zfr[m] = a_frag(sc_iz + m * 16 * SIZ, SIZ, 32, lane);
            }
#pragma unroll 2
            for (int n = 0; n < 8; ++n) {
                v8f acc[2] = {Z8, Z8};
#pragma unroll
                for (int k = 0; k < 4; ++k) {
                    v16h b = b_frag(ws16 + OFF_D1, 128, 32 * k, n * 16, lane);
                    acc[0] = wmma32(ac[0][k], b, acc[0]);
                    acc[1] = wmma32(ac[1][k], b, acc[1]);
                }
                v16h b = b_frag(ws16 + OFF_D1, 128, 128, n * 16, lane);
#pragma unroll
                for (int m = 0; m < 2; ++m) acc[m] = wmma32(zfr[m], b, acc[m]);
#pragma unroll
                for (int m = 0; m < 2; ++m)
                    c_store16(sc_buf + m * 16 * SP, SP, n * 16, lane, acc[m], b_d1, 1);
            }
        }
        __syncthreads();

        // d2: relu(h1 @ d2 + b)
        {
            v16h af[2][4];
#pragma unroll
            for (int m = 0; m < 2; ++m)
#pragma unroll
                for (int k = 0; k < 4; ++k)
                    af[m][k] = a_frag(sc_buf + m * 16 * SP, SP, 32 * k, lane);
#pragma unroll 2
            for (int n = 0; n < 4; ++n) {
                v8f acc[2] = {Z8, Z8};
#pragma unroll
                for (int k = 0; k < 4; ++k) {
                    v16h b = b_frag(ws16 + OFF_D2, 64, 32 * k, n * 16, lane);
                    acc[0] = wmma32(af[0][k], b, acc[0]);
                    acc[1] = wmma32(af[1][k], b, acc[1]);
                }
#pragma unroll
                for (int m = 0; m < 2; ++m)
                    c_store16(sc_h2 + m * 16 * SIZ, SIZ, n * 16, lane, acc[m], b_d2, 1);
            }
        }
        __syncthreads();

        // d3: bbox (N padded to 16) + write output
        {
            v16h h2f[2][2];
#pragma unroll
            for (int m = 0; m < 2; ++m) {
                h2f[m][0] = a_frag(sc_h2 + m * 16 * SIZ, SIZ, 0, lane);
                h2f[m][1] = a_frag(sc_h2 + m * 16 * SIZ, SIZ, 32, lane);
            }
            v8f acc[2] = {Z8, Z8};
            {
                v16h b0 = b_frag(ws16 + OFF_D3P, 16, 0, 0, lane);
                v16h b1 = b_frag(ws16 + OFF_D3P, 16, 32, 0, lane);
#pragma unroll
                for (int m = 0; m < 2; ++m) {
                    acc[m] = wmma32(h2f[m][0], b0, acc[m]);
                    acc[m] = wmma32(h2f[m][1], b1, acc[m]);
                }
            }
            int col = c15;
            float b = (col < 4) ? b_d3[col] : 0.f;
#pragma unroll
            for (int m = 0; m < 2; ++m) {
                h16* bb = sc_bbox + m * 16 * SBB;
#pragma unroll
                for (int v = 0; v < 8; ++v) {
                    float x = acc[m][v] + b;
                    if (col < 4)
                        out[((size_t)(row0 + m * 16 + r0 + v) * 24 + t) * 4 + col] = x;
                    bb[(r0 + v) * SBB + col] = (h16)x;
                }
            }
        }
        __syncthreads();

        // LSTM: z = steplstm[t] + bbox@W4 + prev@U ; fuse gates per 16-col chunk
        {
            v16h pf[2][4], bfr[2];
#pragma unroll
            for (int m = 0; m < 2; ++m) {
#pragma unroll
                for (int k = 0; k < 4; ++k)
                    pf[m][k] = a_frag(sc_prev + m * 16 * SP, SP, 32 * k, lane);
                bfr[m] = a_frag(sc_bbox + m * 16 * SBB, SBB, 0, lane);
            }
#pragma unroll 1
            for (int j = 0; j < 8; ++j) {
                v8f ag[2][4];
#pragma unroll
                for (int g = 0; g < 4; ++g) {
                    int nb = g * 128 + j * 16;
                    v16h b4 = b_frag(ws16 + OFF_L4P, 512, 0, nb, lane);
                    v8f a0 = wmma32(bfr[0], b4, Z8);
                    v8f a1 = wmma32(bfr[1], b4, Z8);
#pragma unroll
                    for (int k = 0; k < 4; ++k) {
                        v16h b = b_frag(ws16 + OFF_U, 512, 32 * k, nb, lane);
                        a0 = wmma32(pf[0][k], b, a0);
                        a1 = wmma32(pf[1][k], b, a1);
                    }
                    ag[0][g] = a0;
                    ag[1][g] = a1;
                }
                int col = j * 16 + c15;
                float bi  = stepl[t * 512 + col];
                float bf_ = stepl[t * 512 + 128 + col];
                float bc  = stepl[t * 512 + 256 + col];
                float bo  = stepl[t * 512 + 384 + col];
#pragma unroll
                for (int m = 0; m < 2; ++m) {
                    h16* pv = sc_prev + m * 16 * SP;
#pragma unroll
                    for (int v = 0; v < 8; ++v) {
                        int lr = r0 + v;
                        float zi  = ag[m][0][v] + bi;
                        float zfv = ag[m][1][v] + bf_;
                        float zc  = ag[m][2][v] + bc;
                        float zo  = ag[m][3][v] + bo;
                        float cp = (float)pv[lr * SP + col];
                        float cn = sigm(zfv) * cp + sigm(zi) * tanhf(zc);
                        float hn = sigm(zo) * tanhf(cn);
                        pv[lr * SP + col] = (h16)hn;
                    }
                }
            }
        }
        __syncthreads();
    }
}

// ---------------- host side ----------------
extern "C" void kernel_launch(void* const* d_in, const int* in_sizes, int n_in,
                              void* d_out, int out_size, void* d_ws, size_t ws_size,
                              hipStream_t stream) {
    (void)n_in; (void)out_size; (void)ws_size;
    const float* label_set = (const float*)d_in[0];
    const float* bbox_in   = (const float*)d_in[1];
    const float* eps       = (const float*)d_in[2];
    const float* c_ls1_W = (const float*)d_in[3];
    const float* c_ls1_b = (const float*)d_in[4];
    const float* c_ls2_W = (const float*)d_in[5];
    const float* c_ls2_b = (const float*)d_in[6];
    const float* c_cl1_W = (const float*)d_in[7];
    const float* c_cl1_b = (const float*)d_in[8];
    const float* c_cl2_W = (const float*)d_in[9];
    const float* c_cl2_b = (const float*)d_in[10];
    const float* c_pb_W  = (const float*)d_in[11];
    const float* c_pb_b  = (const float*)d_in[12];
    const float* c_out_W = (const float*)d_in[13];
    const float* c_out_b = (const float*)d_in[14];
    const float* e1_W = (const float*)d_in[15];
    const float* e1_b = (const float*)d_in[16];
    const float* e2_W = (const float*)d_in[17];
    const float* e2_b = (const float*)d_in[18];
    const float* e3_W = (const float*)d_in[19];
    const float* e3_b = (const float*)d_in[20];
    const float* e_mu_W = (const float*)d_in[21];
    const float* e_mu_b = (const float*)d_in[22];
    const float* e_lv_W = (const float*)d_in[23];
    const float* e_lv_b = (const float*)d_in[24];
    const float* d1_W = (const float*)d_in[25];
    const float* d1_b = (const float*)d_in[26];
    const float* d2_W = (const float*)d_in[27];
    const float* d2_b = (const float*)d_in[28];
    const float* d3_W = (const float*)d_in[29];
    const float* d3_b = (const float*)d_in[30];
    const float* lstm_W = (const float*)d_in[37];
    const float* lstm_U = (const float*)d_in[38];
    const float* lstm_b = (const float*)d_in[39];

    h16*   ws16  = (h16*)d_ws;
    float* stepc = (float*)((char*)d_ws + OFF_STEPC_BYTES);
    float* stepl = (float*)((char*)d_ws + OFF_STEPL_BYTES);
    float* zb    = (float*)((char*)d_ws + OFF_ZB_BYTES);
    float* out   = (float*)d_out;
    const int B = in_sizes[0] / N_LABELS;

    // zero f16 region (covers all K/N zero-padding)
    zero16_kernel<<<(TOT_H16 + 255) / 256, 256, 0, stream>>>(ws16, TOT_H16);

    auto cvt = [&](int dstoff, int ld_dst, int dr0, int dc0, const float* src,
                   int ld_src, int sr0, int sc0, int rows, int cols) {
        int n = rows * cols;
        conv_kernel<<<(n + 255) / 256, 256, 0, stream>>>(ws16 + dstoff, ld_dst, dr0, dc0,
                                                         src, ld_src, sr0, sc0, rows, cols);
    };
    cvt(OFF_CPB, 128, 0, 0, c_pb_W, 128, 0, 0, 128, 128);
    cvt(OFF_C3, 128, 0, 0, c_out_W, 128, 256, 0, 128, 128);
    cvt(OFF_E1P, 128, 0, 0, e1_W, 128, 0, 0, 4, 128);
    cvt(OFF_E2, 128, 0, 0, e2_W, 128, 0, 0, 128, 128);
    cvt(OFF_E3, 32, 0, 0, e3_W, 32, 0, 0, 256, 32);
    cvt(OFF_ML, 64, 0, 0, e_mu_W, 32, 0, 0, 32, 32);
    cvt(OFF_ML, 64, 0, 32, e_lv_W, 32, 0, 0, 32, 32);
    cvt(OFF_D1, 128, 0, 0, d1_W, 128, 0, 0, 160, 128);
    cvt(OFF_D2, 64, 0, 0, d2_W, 64, 0, 0, 128, 64);
    cvt(OFF_D3P, 16, 0, 0, d3_W, 4, 0, 0, 64, 4);
    cvt(OFF_L4P, 512, 0, 0, lstm_W, 512, 24, 0, 4, 512);
    cvt(OFF_U, 512, 0, 0, lstm_U, 512, 0, 0, 128, 512);
    cvt(OFF_LS1P, 128, 0, 0, c_ls1_W, 128, 0, 0, 24, 128);
    cvt(OFF_LS2, 128, 0, 0, c_ls2_W, 128, 0, 0, 128, 128);
    cvt(OFF_CA, 128, 0, 0, c_out_W, 128, 0, 0, 128, 128);

    step_consts_kernel<<<NSTEP, 128, 0, stream>>>(c_cl1_W, c_cl1_b, c_cl2_W, c_cl2_b,
                                                  c_out_W, c_out_b, lstm_W, lstm_b,
                                                  stepc, stepl, zb);

    bboxvae_main<<<B / 32, 32, 0, stream>>>(label_set, bbox_in, eps, ws16, stepc, stepl,
                                            zb, c_ls1_b, c_ls2_b, c_pb_b, e1_b, e2_b,
                                            e3_b, e_mu_b, e_lv_b, d1_b, d2_b, d3_b, out);
}
